// GLF_25426206392719
// MI455X (gfx1250) — matrix operand order repaired and checked by reading, not compiled, not run on hardware
//
#include <hip/hip_runtime.h>

// ---------------------------------------------------------------------------
// Problem constants (from the reference)
// ---------------------------------------------------------------------------
constexpr int E     = 1024;
constexpr int HEADS = 16;
constexpr int DH    = 64;              // E / HEADS
constexpr int BATCH = 2;
constexpr int SEQ   = 2048;
constexpr int MTOK  = BATCH * SEQ;     // 4096 tokens
constexpr int E2    = 2 * E;
constexpr int E3    = 3 * E;

typedef __bf16 bf16_t;
typedef __attribute__((ext_vector_type(16))) __bf16 v16bf;
typedef __attribute__((ext_vector_type(8)))  float  v8f;
typedef __attribute__((ext_vector_type(4)))  unsigned int u32x4;
typedef __attribute__((ext_vector_type(8)))  int          i32x8;
typedef __attribute__((ext_vector_type(4)))  int          i32x4;

union ABFrag {
  v16bf  v;
  uint4  q[2];
  bf16_t h[16];
};

__device__ __forceinline__ v8f wmma_bf16(v16bf a, v16bf b, v8f c) {
  // D = A(16x32 bf16) * B(32x16 bf16) + C(16x16 f32)
  return __builtin_amdgcn_wmma_f32_16x16x32_bf16(false, a, false, b,
                                                 (short)0, c, false, false);
}

// Reductions confined to a 16-lane half of the wave (xor masks < 16).
__device__ __forceinline__ float hmax16(float x) {
  for (int d = 1; d < 16; d <<= 1) x = fmaxf(x, __shfl_xor(x, d, 32));
  return x;
}
__device__ __forceinline__ float hsum16(float x) {
  for (int d = 1; d < 16; d <<= 1) x += __shfl_xor(x, d, 32);
  return x;
}

// ---------------------------------------------------------------------------
// fp32 -> bf16 conversion (one-time; amortized over M=4096 reuses)
// ---------------------------------------------------------------------------
__global__ void cvt_f32_bf16(const float* __restrict__ src,
                             bf16_t* __restrict__ dst, int n4) {
  int i = blockIdx.x * blockDim.x + threadIdx.x;
  if (i < n4) {
    float4 f = ((const float4*)src)[i];
    bf16_t* d = dst + (size_t)i * 4;
    d[0] = (bf16_t)f.x; d[1] = (bf16_t)f.y;
    d[2] = (bf16_t)f.z; d[3] = (bf16_t)f.w;
  }
}

// ---------------------------------------------------------------------------
// WMMA GEMM with TDM-staged weight tiles:
//   C[M,N] = A[M,K](bf16,rm) * W[N,K](bf16,rm)^T + bias[N]
// Block: 256 threads = 8 waves -> 128(row) x 64(col) macro-tile.
// Per K-step (32): wave0 issues a Tensor-Data-Mover load of the 64x32 bf16
// W-tile into LDS (double-buffered, TENSORcnt), every wave reuses one
// A-fragment across 4 column-tile WMMAs.
// ---------------------------------------------------------------------------
template <typename OutT>
__global__ __launch_bounds__(256)
void gemm_bf16_wt(const bf16_t* __restrict__ A, int lda,
                  const bf16_t* __restrict__ W, int ldw,
                  const float* __restrict__ bias,
                  OutT* __restrict__ C, int ldc, int K, int Ntot) {
  __shared__ bf16_t Wt[2][64][32];     // [buf][col-in-tile][k-in-step]

  const int lane = threadIdx.x & 31;
  const int wave = threadIdx.x >> 5;
  const int nl   = lane & 15;
  const int hl   = lane >> 4;
  const int rowBase = blockIdx.x * 128 + wave * 16;
  const int colBase = blockIdx.y * 64;

  const bf16_t* arow = A + (size_t)(rowBase + nl) * lda;

  const uint32_t ldsoff[2] = { (uint32_t)(uintptr_t)&Wt[0][0][0],
                               (uint32_t)(uintptr_t)&Wt[1][0][0] };

  // Issue one TDM descriptor: 2-D tensor (dim0=K, dim1=N, stride=ldw),
  // tile 32(K) x 64(cols), 2-byte elements, into LDS buffer `buf`.
  auto tdm_issue = [&](int buf, int kk) {
    if (wave == 0) {
      uint64_t ga = (uint64_t)(uintptr_t)(W + (size_t)colBase * ldw + kk);
      u32x4 g0;
      g0[0] = 1u;                                   // count=1, rest of flags 0
      g0[1] = ldsoff[buf];                          // lds_addr
      g0[2] = (uint32_t)ga;                         // global_addr[31:0]
      g0[3] = (uint32_t)((ga >> 32) & 0x1FFFFFFu)   // global_addr[56:32]
              | (2u << 30);                         // type = 2 ("image")
      uint32_t d0 = (uint32_t)K;                    // tensor_dim0
      uint32_t d1 = (uint32_t)Ntot;                 // tensor_dim1
      uint64_t s0 = (uint64_t)ldw;                  // tensor_dim0_stride
      i32x8 g1;
      g1[0] = (int)(1u << 16);                      // data_size=1 -> 2 bytes
      g1[1] = (int)((d0 & 0xFFFFu) << 16);          // dim0[15:0] | bar_addr=0
      g1[2] = (int)(((d1 & 0xFFFFu) << 16) | (d0 >> 16));
      g1[3] = (int)((32u << 16) | (d1 >> 16));      // tile_dim0=32
      g1[4] = (int)64;                              // tile_dim1=64, tile_dim2=0
      g1[5] = (int)(uint32_t)(s0 & 0xFFFFFFFFu);    // dim0_stride[31:0]
      g1[6] = (int)(uint32_t)((s0 >> 32) & 0xFFFFu);// dim0_stride[47:32]
      g1[7] = 0;
      i32x4 gz4 = {0, 0, 0, 0};
      i32x8 gz8 = {0, 0, 0, 0, 0, 0, 0, 0};
      // clang-23 / therock signature: (g0, g1, g2, g3, g4, cpol)
      __builtin_amdgcn_tensor_load_to_lds(g0, g1, gz4, gz4, gz8, 0);
    }
  };

  v8f acc[4] = {};
  tdm_issue(0, 0);
  for (int kk = 0; kk < K; kk += 32) {
    const int cur = (kk >> 5) & 1;
    const bool more = (kk + 32) < K;
    if (more) {
      tdm_issue(1 - cur, kk + 32);
      __builtin_amdgcn_s_wait_tensorcnt(1);   // current tile landed
    } else {
      __builtin_amdgcn_s_wait_tensorcnt(0);
    }
    __syncthreads();

    ABFrag a;   // A-frag: row nl; elems 0..15 = K kk+hl*16..+15
    a.q[0] = *(const uint4*)(arow + kk + hl * 16);
    a.q[1] = *(const uint4*)(arow + kk + hl * 16 + 8);
    __builtin_prefetch(arow + kk + 64, 0, 1);   // global_prefetch_b8

    for (int t = 0; t < 4; ++t) {
      ABFrag b; // B-frag (W^T): lane col t*16+nl; elems = K hl*16..+15
      b.q[0] = *(const uint4*)(&Wt[cur][t * 16 + nl][hl * 16]);
      b.q[1] = *(const uint4*)(&Wt[cur][t * 16 + nl][hl * 16 + 8]);
      acc[t] = wmma_bf16(a.v, b.v, acc[t]);
    }
    __syncthreads();   // done with Wt[cur] before it is overwritten
  }

  for (int t = 0; t < 4; ++t) {
    const int col = colBase + t * 16 + nl;
    const float bv = bias ? bias[col] : 0.0f;
    OutT* cp = C + (size_t)(rowBase + 8 * hl) * ldc + col;
    for (int j = 0; j < 8; ++j)                 // C: elem j -> row j + 8*hl
      cp[(size_t)j * ldc] = (OutT)(acc[t][j] + bv);
  }
}

// ---------------------------------------------------------------------------
// Global attention, flash-style. One wave per (batch, head, 16-row Q tile).
// qkv layout per token row (stride 3E): [Q | K | V], head h at col h*DH.
// V chunks are staged LDS-ward with async-to-LDS loads (ASYNCcnt).
// ---------------------------------------------------------------------------
__global__ __launch_bounds__(32)
void flash_attn_global(const bf16_t* __restrict__ qkv,
                       bf16_t* __restrict__ out) {
  const int b  = blockIdx.x / HEADS;
  const int h  = blockIdx.x % HEADS;
  const int qt = blockIdx.y;
  const int lane = threadIdx.x & 31;
  const int nl = lane & 15, hl = lane >> 4;

  const int tokQ = b * SEQ + qt * 16;
  const bf16_t* qbase = qkv + (size_t)(tokQ + nl) * E3 + h * DH;

  // Q fragments for the two dh slices (K = 0..31, 32..63)
  ABFrag aq0, aq1;
  aq0.q[0] = *(const uint4*)(qbase + hl * 16);
  aq0.q[1] = *(const uint4*)(qbase + hl * 16 + 8);
  aq1.q[0] = *(const uint4*)(qbase + 32 + hl * 16);
  aq1.q[1] = *(const uint4*)(qbase + 32 + hl * 16 + 8);

  __shared__ bf16_t Plds[16][32];   // probs tile, C-layout -> A-layout transpose
  __shared__ bf16_t Vlds[32][64];   // staged V chunk [key][dh]

  float rowmax[8], rowsum[8];
  v8f o[4] = {};
  for (int j = 0; j < 8; ++j) { rowmax[j] = -1e30f; rowsum[j] = 0.0f; }

  const float scale = 0.125f;       // dh^-0.5, dh = 64
  const bf16_t* kvbase = qkv + (size_t)(b * SEQ) * E3;

  for (int n0 = 0; n0 < SEQ; n0 += 32) {
    // kick off async V staging for this chunk: lane copies 64 B (4 x b128) of its V row
    {
      const bf16_t* vrow = kvbase + (size_t)(n0 + lane) * E3 + 2 * E + h * DH;
      uint32_t ldsa = (uint32_t)(uintptr_t)(&Vlds[lane][0]);
      asm volatile(
          "global_load_async_to_lds_b128 %0, %1, off\n\t"
          "global_load_async_to_lds_b128 %0, %1, off offset:16\n\t"
          "global_load_async_to_lds_b128 %0, %1, off offset:32\n\t"
          "global_load_async_to_lds_b128 %0, %1, off offset:48"
          :: "v"(ldsa), "v"(vrow) : "memory");
    }

    // --- scores: two 16x16 tiles over this 32-key chunk -------------------
    v8f sc[2];
    for (int t = 0; t < 2; ++t) {
      const bf16_t* krow = kvbase + (size_t)(n0 + t * 16 + nl) * E3 + E + h * DH;
      ABFrag bk0, bk1;   // B = K^T: elems 0..15 = dh hl*16..+15 (per slice)
      bk0.q[0] = *(const uint4*)(krow + hl * 16);
      bk0.q[1] = *(const uint4*)(krow + hl * 16 + 8);
      bk1.q[0] = *(const uint4*)(krow + 32 + hl * 16);
      bk1.q[1] = *(const uint4*)(krow + 32 + hl * 16 + 8);
      v8f c = {};
      c = wmma_bf16(aq0.v, bk0.v, c);
      c = wmma_bf16(aq1.v, bk1.v, c);
      sc[t] = c;
    }

    // --- online softmax update -------------------------------------------
    float alpha[8], nm[8];
    for (int j = 0; j < 8; ++j) {
      sc[0][j] *= scale; sc[1][j] *= scale;
      float mx = hmax16(fmaxf(sc[0][j], sc[1][j]));
      nm[j] = fmaxf(rowmax[j], mx);
      alpha[j] = __expf(rowmax[j] - nm[j]);
      rowmax[j] = nm[j];
    }
    for (int j = 0; j < 8; ++j) {
      sc[0][j] = __expf(sc[0][j] - nm[j]);
      sc[1][j] = __expf(sc[1][j] - nm[j]);
      rowsum[j] = rowsum[j] * alpha[j] + hsum16(sc[0][j] + sc[1][j]);
    }

    // --- transpose P through LDS into A-fragment layout -------------------
    for (int j = 0; j < 8; ++j) {
      const int r = j + 8 * hl;              // C-layout row
      Plds[r][nl]      = (bf16_t)sc[0][j];
      Plds[r][nl + 16] = (bf16_t)sc[1][j];
    }
    asm volatile("s_wait_asynccnt 0x0" ::: "memory");   // V chunk landed
    __syncthreads();

    ABFrag pf;                                // P as A: row nl, keys hl*16..+15
    pf.q[0] = *(const uint4*)(&Plds[nl][hl * 16]);
    pf.q[1] = *(const uint4*)(&Plds[nl][hl * 16 + 8]);

    for (int t = 0; t < 4; ++t) {
      for (int j = 0; j < 8; ++j) o[t][j] *= alpha[j];
      ABFrag bv;                              // V as B: elem e = key hl*16+e
      for (int e = 0; e < 16; ++e)
        bv.h[e] = Vlds[hl * 16 + e][t * 16 + nl];
      o[t] = wmma_bf16(pf.v, bv.v, o[t]);
    }
    __syncthreads();
  }

  // normalize + store (bf16, [MTOK, E])
  bf16_t* ob = out + (size_t)(tokQ + 8 * hl) * E + h * DH;
  for (int t = 0; t < 4; ++t)
    for (int j = 0; j < 8; ++j)
      ob[(size_t)j * E + t * 16 + nl] = (bf16_t)(o[t][j] / rowsum[j]);
}

// ---------------------------------------------------------------------------
// Block-local attention: 16-token blocks, one wave per (block, head).
// Single 16-key chunk, zero-padded to K=32 for the PV WMMA.
// ---------------------------------------------------------------------------
__global__ __launch_bounds__(32)
void local_attn(const bf16_t* __restrict__ qkv, bf16_t* __restrict__ out) {
  const int h    = blockIdx.x % HEADS;
  const int blk  = blockIdx.x / HEADS;
  const int tok0 = blk * 16;
  const int lane = threadIdx.x & 31;
  const int nl = lane & 15, hl = lane >> 4;

  const bf16_t* qbase = qkv + (size_t)(tok0 + nl) * E3 + h * DH;
  ABFrag aq0, aq1;
  aq0.q[0] = *(const uint4*)(qbase + hl * 16);
  aq0.q[1] = *(const uint4*)(qbase + hl * 16 + 8);
  aq1.q[0] = *(const uint4*)(qbase + 32 + hl * 16);
  aq1.q[1] = *(const uint4*)(qbase + 32 + hl * 16 + 8);

  const bf16_t* krow = qkv + (size_t)(tok0 + nl) * E3 + E + h * DH;
  ABFrag bk0, bk1;
  bk0.q[0] = *(const uint4*)(krow + hl * 16);
  bk0.q[1] = *(const uint4*)(krow + hl * 16 + 8);
  bk1.q[0] = *(const uint4*)(krow + 32 + hl * 16);
  bk1.q[1] = *(const uint4*)(krow + 32 + hl * 16 + 8);

  v8f c = {};
  c = wmma_bf16(aq0.v, bk0.v, c);
  c = wmma_bf16(aq1.v, bk1.v, c);

  const float scale = 0.125f;
  float rowsum[8];
  __shared__ bf16_t Plds[16][32];
  __shared__ bf16_t Vlds[32][64];

  for (int j = 0; j < 8; ++j) {
    float s  = c[j] * scale;
    float mx = hmax16(s);
    float p  = __expf(s - mx);
    rowsum[j] = hsum16(p);
    const int r = j + 8 * hl;
    Plds[r][nl]      = (bf16_t)p;
    Plds[r][nl + 16] = (bf16_t)0.0f;          // zero-pad keys 16..31
  }
  // stage V: rows 0..15 real, rows 16..31 zero (safe padding for WMMA)
  {
    uint4* dv = (uint4*)(&Vlds[lane][0]);
    if (lane < 16) {
      const uint4* sv = (const uint4*)(qkv + (size_t)(tok0 + lane) * E3 + 2 * E + h * DH);
      dv[0] = sv[0]; dv[1] = sv[1]; dv[2] = sv[2]; dv[3] = sv[3];
    } else {
      uint4 z = {0, 0, 0, 0};
      dv[0] = z; dv[1] = z; dv[2] = z; dv[3] = z;
    }
  }
  __syncthreads();

  ABFrag pf;
  pf.q[0] = *(const uint4*)(&Plds[nl][hl * 16]);
  pf.q[1] = *(const uint4*)(&Plds[nl][hl * 16 + 8]);

  bf16_t* ob = out + (size_t)(tok0 + 8 * hl) * E + h * DH;
  for (int t = 0; t < 4; ++t) {
    ABFrag bv;
    for (int e = 0; e < 16; ++e)
      bv.h[e] = Vlds[hl * 16 + e][t * 16 + nl];
    v8f o = {};
    o = wmma_bf16(pf.v, bv.v, o);
    for (int j = 0; j < 8; ++j)
      ob[(size_t)j * E + t * 16 + nl] = (bf16_t)(o[j] / rowsum[j]);
  }
}

// ---------------------------------------------------------------------------
// Host orchestration
// ---------------------------------------------------------------------------
extern "C" void kernel_launch(void* const* d_in, const int* in_sizes, int n_in,
                              void* d_out, int out_size, void* d_ws, size_t ws_size,
                              hipStream_t stream) {
  (void)in_sizes; (void)n_in; (void)out_size; (void)ws_size;

  const float* x       = (const float*)d_in[0];
  const float* w_in_g  = (const float*)d_in[1];
  const float* b_in_g  = (const float*)d_in[2];
  const float* w_out_g = (const float*)d_in[3];
  const float* b_out_g = (const float*)d_in[4];
  const float* w_in_l  = (const float*)d_in[5];
  const float* b_in_l  = (const float*)d_in[6];
  const float* w_out_l = (const float*)d_in[7];
  const float* b_out_l = (const float*)d_in[8];
  const float* w_f     = (const float*)d_in[9];
  const float* b_f     = (const float*)d_in[10];
  float* out = (float*)d_out;

  char* ws = (char*)d_ws;
  size_t off = 0;
  auto carve = [&](size_t bytes) -> char* {
    char* p = ws + off;
    off = (off + bytes + 255) & ~(size_t)255;
    return p;
  };
  bf16_t* xb    = (bf16_t*)carve((size_t)MTOK * E  * 2);
  bf16_t* wing  = (bf16_t*)carve((size_t)E3   * E  * 2);
  bf16_t* winl  = (bf16_t*)carve((size_t)E3   * E  * 2);
  bf16_t* woutg = (bf16_t*)carve((size_t)E    * E  * 2);
  bf16_t* woutl = (bf16_t*)carve((size_t)E    * E  * 2);
  bf16_t* wf    = (bf16_t*)carve((size_t)E    * E2 * 2);
  bf16_t* qkvg  = (bf16_t*)carve((size_t)MTOK * E3 * 2);
  bf16_t* qkvl  = (bf16_t*)carve((size_t)MTOK * E3 * 2);
  bf16_t* attng = (bf16_t*)carve((size_t)MTOK * E  * 2);
  bf16_t* attnl = (bf16_t*)carve((size_t)MTOK * E  * 2);
  bf16_t* fused = (bf16_t*)carve((size_t)MTOK * E2 * 2);

  auto cvt = [&](const float* s, bf16_t* d, size_t n) {
    int n4 = (int)(n / 4);
    cvt_f32_bf16<<<dim3((n4 + 255) / 256), dim3(256), 0, stream>>>(s, d, n4);
  };
  cvt(x,       xb,    (size_t)MTOK * E);
  cvt(w_in_g,  wing,  (size_t)E3 * E);
  cvt(w_in_l,  winl,  (size_t)E3 * E);
  cvt(w_out_g, woutg, (size_t)E * E);
  cvt(w_out_l, woutl, (size_t)E * E);
  cvt(w_f,     wf,    (size_t)E * E2);

  // QKV projections (global + local)
  gemm_bf16_wt<bf16_t><<<dim3(MTOK / 128, E3 / 64), 256, 0, stream>>>(
      xb, E, wing, E, b_in_g, qkvg, E3, E, E3);
  gemm_bf16_wt<bf16_t><<<dim3(MTOK / 128, E3 / 64), 256, 0, stream>>>(
      xb, E, winl, E, b_in_l, qkvl, E3, E, E3);

  // Attention cores
  flash_attn_global<<<dim3(BATCH * HEADS, SEQ / 16), 32, 0, stream>>>(qkvg, attng);
  local_attn<<<dim3((MTOK / 16) * HEADS), 32, 0, stream>>>(qkvl, attnl);

  // Output projections, written into the concatenated buffer (ldc = 2E)
  gemm_bf16_wt<bf16_t><<<dim3(MTOK / 128, E / 64), 256, 0, stream>>>(
      attng, E, woutg, E, b_out_g, fused, E2, E, E);
  gemm_bf16_wt<bf16_t><<<dim3(MTOK / 128, E / 64), 256, 0, stream>>>(
      attnl, E, woutl, E, b_out_l, fused + E, E2, E, E);

  // Final fuse: out[M,E] = fused[M,2E] @ w_f[E,2E]^T + b_f   (fp32 out)
  gemm_bf16_wt<float><<<dim3(MTOK / 128, E / 64), 256, 0, stream>>>(
      fused, E2, wf, E2, b_f, out, E, E2, E);
}
